// GraphNetwork_9938554323103
// MI455X (gfx1250) — matrix operand order repaired and checked by reading, not compiled, not run on hardware
//
#include <hip/hip_runtime.h>

typedef __attribute__((ext_vector_type(8)))  __bf16 v8bf;
typedef __attribute__((ext_vector_type(16))) __bf16 v16bf;
typedef __attribute__((ext_vector_type(8)))  float  v8f;

typedef unsigned short u16;
typedef unsigned int   u32;
typedef unsigned long long u64;

__device__ __forceinline__ u16 f32_to_bf16(float f) {
  u32 u = __float_as_uint(f);
  u32 r = u + 0x7FFFu + ((u >> 16) & 1u);   // round-to-nearest-even
  return (u16)(r >> 16);
}
__device__ __forceinline__ float bf16_to_f32(u16 h) {
  return __uint_as_float(((u32)h) << 16);
}

// ---------------------------------------------------------------------------
// Weight prep: Wt[o][k] = bf16( k < Fin ? ws[k][o] : wn[k-Fin][o] )
// Row-major [Fout][2*Fin]: per output column, K is contiguous.
// ---------------------------------------------------------------------------
__global__ void prep_weights(const float* __restrict__ ws,
                             const float* __restrict__ wn,
                             u16* __restrict__ wt, int Fin, int Fout) {
  int idx = blockIdx.x * blockDim.x + threadIdx.x;
  int K2 = 2 * Fin;
  if (idx >= Fout * K2) return;
  int o = idx / K2, k = idx % K2;
  float v = (k < Fin) ? ws[(size_t)k * Fout + o] : wn[(size_t)(k - Fin) * Fout + o];
  wt[(size_t)o * K2 + k] = f32_to_bf16(v);
}

// ---------------------------------------------------------------------------
// Input prep: inputs [B][N][F] f32  ->  Z[:,0:F] bf16 with row m = n*B+b
// ---------------------------------------------------------------------------
__global__ void prep_input(const float* __restrict__ in, u16* __restrict__ Z,
                           int N, int B, int F, int ld) {
  size_t idx = (size_t)blockIdx.x * blockDim.x + threadIdx.x;
  size_t total = (size_t)N * B * F;
  if (idx >= total) return;
  int f = (int)(idx % F);
  size_t t = idx / F;
  int n = (int)(t % N);
  int b = (int)(t / N);
  Z[((size_t)n * B + b) * ld + f] = f32_to_bf16(in[idx]);
}

// ---------------------------------------------------------------------------
// Aggregation: Z[m, F+f] = 0.25 * sum_{i<4} Z[src[4n+i]*B + b, f]
// 8 bf16 features per thread (uint4 loads/stores).
// ---------------------------------------------------------------------------
__global__ void agg_kernel(u16* __restrict__ Z, const int* __restrict__ src,
                           int N, int B, int F, int ld) {
  size_t idx = (size_t)blockIdx.x * blockDim.x + threadIdx.x;
  int fb = F >> 3;
  size_t total = (size_t)N * B * fb;
  if (idx >= total) return;
  int f8 = (int)(idx % fb);
  size_t t = idx / fb;
  int b = (int)(t % B);
  int n = (int)(t / B);

  float s[8];
#pragma unroll
  for (int r = 0; r < 8; ++r) s[r] = 0.0f;

#pragma unroll
  for (int i = 0; i < 4; ++i) {
    int sn = src[n * 4 + i];
    const uint4* p = reinterpret_cast<const uint4*>(
        Z + ((size_t)sn * B + b) * ld + (size_t)f8 * 8);
    uint4 v = *p;
    u32 w[4] = {v.x, v.y, v.z, v.w};
#pragma unroll
    for (int j = 0; j < 4; ++j) {
      s[2 * j]     += bf16_to_f32((u16)(w[j] & 0xFFFFu));
      s[2 * j + 1] += bf16_to_f32((u16)(w[j] >> 16));
    }
  }
  u32 o[4];
#pragma unroll
  for (int j = 0; j < 4; ++j) {
    u32 lo = f32_to_bf16(0.25f * s[2 * j]);
    u32 hi = f32_to_bf16(0.25f * s[2 * j + 1]);
    o[j] = lo | (hi << 16);
  }
  uint4 ov = {o[0], o[1], o[2], o[3]};
  *reinterpret_cast<uint4*>(Z + ((size_t)n * B + b) * ld + F + (size_t)f8 * 8) = ov;
}

// ---------------------------------------------------------------------------
// Column-block copy for the concat skip connections.
// ---------------------------------------------------------------------------
__global__ void copy_cols(const u16* __restrict__ Zs, int lds_,
                          u16* __restrict__ Zd, int ldd, int coff,
                          int M, int Fc) {
  size_t idx = (size_t)blockIdx.x * blockDim.x + threadIdx.x;
  int fb = Fc >> 3;
  size_t total = (size_t)M * fb;
  if (idx >= total) return;
  int f8 = (int)(idx % fb);
  int m = (int)(idx / fb);
  const uint4* s = reinterpret_cast<const uint4*>(Zs + (size_t)m * lds_ + (size_t)f8 * 8);
  *reinterpret_cast<uint4*>(Zd + (size_t)m * ldd + coff + (size_t)f8 * 8) = *s;
}

// ---------------------------------------------------------------------------
// Fused SAGE GEMM:  Y[M][Fout] = Z[M][K] * Wt^T + bias  (bf16 in, f32 acc)
//
// Block = 256 threads = 8 waves, covering 128 rows x 64 cols of output.
// Phase 1: async-stage the block's 64-col x K weight slab into LDS
//          (global_load_async_to_lds_b128, ASYNCcnt, VGPR-free) with a padded
//          row stride P = K+8 elements so half-wave fragment reads hit 16
//          distinct 4-dword bank groups.
// Phase 2: K-loop, step 32: A fragments from global (2x b128/lane), B
//          fragments from LDS (ds pipe), 4x v_wmma_f32_16x16x32_bf16.
//
// Fragment maps (wave32):
//   A 16x32: lane<16 row=lane  K={k..k+7, k+16..k+23}; lane>=16 row=lane-16,
//            K={k+8..k+15, k+24..k+31}
//   B 32x16: lane<16 col=lane  K={k..k+7, k+16..k+23}; lane>=16 col=lane-16 K={k+8..k+15, k+24..k+31}
//   C/D f32: col = lane%16; rows (lane/16)*8 + r  for vgpr r=0..7
// ---------------------------------------------------------------------------
extern __shared__ u16 smem[];

__global__ void sage_gemm(const u16* __restrict__ Z, const u16* __restrict__ Wt,
                          const float* __restrict__ bias,
                          u16* __restrict__ dstZ, int ldd, int colOff,
                          int M, int K, int Fout, int relu,
                          float* __restrict__ dstF, int N, int B) {
  const int lane = threadIdx.x & 31;
  const int wave = threadIdx.x >> 5;
  const int m0 = (blockIdx.x * 8 + wave) * 16;
  const int n0 = blockIdx.y * 64;            // block's 64-wide column slab
  const int P  = K + 8;                      // padded LDS row stride (elements)

  // -------- Phase 1: cooperative async stage of Wt[n0:n0+64, 0:K] ----------
  {
    const u32 ldsBase = __builtin_amdgcn_groupstaticsize();  // dyn-LDS start
    const int nchunk = (64 * K) / 8;         // 16B chunks; multiple of 256
    for (int chunk = threadIdx.x; chunk < nchunk; chunk += 256) {
      int c  = chunk / (K / 8);
      int kk = (chunk - c * (K / 8)) * 8;
      u64 ga = (u64)(Wt + (size_t)(n0 + c) * K + kk);
      u32 la = ldsBase + (u32)((c * P + kk) * 2);
      asm volatile("global_load_async_to_lds_b128 %0, %1, off"
                   :: "v"(la), "v"(ga) : "memory");
    }
    asm volatile("s_wait_asynccnt 0x0" ::: "memory");
  }
  __syncthreads();

  // -------- Phase 2: WMMA main loop ----------------------------------------
  const int lh  = lane >> 4;                 // 0 or 1
  const int l16 = lane & 15;

  const u16* arow = Z + (size_t)(m0 + l16) * K + lh * 8;
  const u16* b0p = smem + (size_t)(0 * 16 + l16) * P + lh * 8;
  const u16* b1p = smem + (size_t)(1 * 16 + l16) * P + lh * 8;
  const u16* b2p = smem + (size_t)(2 * 16 + l16) * P + lh * 8;
  const u16* b3p = smem + (size_t)(3 * 16 + l16) * P + lh * 8;

  v8f acc[4];
#pragma unroll
  for (int j = 0; j < 4; ++j) acc[j] = (v8f){0.f, 0.f, 0.f, 0.f, 0.f, 0.f, 0.f, 0.f};

#pragma unroll 4
  for (int k = 0; k < K; k += 32) {
    v8bf alo = *reinterpret_cast<const v8bf*>(arow + k);
    v8bf ahi = *reinterpret_cast<const v8bf*>(arow + k + 16);
    v16bf a = __builtin_shufflevector(alo, ahi,
                                      0, 1, 2, 3, 4, 5, 6, 7,
                                      8, 9, 10, 11, 12, 13, 14, 15);
    v8bf b0l = *reinterpret_cast<const v8bf*>(b0p + k);
    v8bf b0h = *reinterpret_cast<const v8bf*>(b0p + k + 16);
    v8bf b1l = *reinterpret_cast<const v8bf*>(b1p + k);
    v8bf b1h = *reinterpret_cast<const v8bf*>(b1p + k + 16);
    v8bf b2l = *reinterpret_cast<const v8bf*>(b2p + k);
    v8bf b2h = *reinterpret_cast<const v8bf*>(b2p + k + 16);
    v8bf b3l = *reinterpret_cast<const v8bf*>(b3p + k);
    v8bf b3h = *reinterpret_cast<const v8bf*>(b3p + k + 16);
    v16bf b0 = __builtin_shufflevector(b0l, b0h, 0,1,2,3,4,5,6,7,8,9,10,11,12,13,14,15);
    v16bf b1 = __builtin_shufflevector(b1l, b1h, 0,1,2,3,4,5,6,7,8,9,10,11,12,13,14,15);
    v16bf b2 = __builtin_shufflevector(b2l, b2h, 0,1,2,3,4,5,6,7,8,9,10,11,12,13,14,15);
    v16bf b3 = __builtin_shufflevector(b3l, b3h, 0,1,2,3,4,5,6,7,8,9,10,11,12,13,14,15);
    acc[0] = __builtin_amdgcn_wmma_f32_16x16x32_bf16(false, a, false, b0,
                                                     (short)0, acc[0], false, false);
    acc[1] = __builtin_amdgcn_wmma_f32_16x16x32_bf16(false, a, false, b1,
                                                     (short)0, acc[1], false, false);
    acc[2] = __builtin_amdgcn_wmma_f32_16x16x32_bf16(false, a, false, b2,
                                                     (short)0, acc[2], false, false);
    acc[3] = __builtin_amdgcn_wmma_f32_16x16x32_bf16(false, a, false, b3,
                                                     (short)0, acc[3], false, false);
  }

  // -------- Epilogue: bias + ReLU, bf16 store (+ optional f32 out) ---------
#pragma unroll
  for (int j = 0; j < 4; ++j) {
    int col = n0 + j * 16 + l16;
    float bs = bias[col];
#pragma unroll
    for (int r = 0; r < 8; ++r) {
      float v = acc[j][r] + bs;
      if (relu) v = fmaxf(v, 0.0f);
      int m = m0 + lh * 8 + r;
      dstZ[(size_t)m * ldd + colOff + col] = f32_to_bf16(v);
      if (dstF) {
        int node = m / B;
        int b = m - node * B;
        dstF[((size_t)b * N + node) * Fout + col] = v;  // [B][N][Fout]
      }
    }
  }
}

// ---------------------------------------------------------------------------
// Host orchestration (all launches on `stream`, graph-capture safe).
// ---------------------------------------------------------------------------
extern "C" void kernel_launch(void* const* d_in, const int* in_sizes, int n_in,
                              void* d_out, int out_size, void* d_ws, size_t ws_size,
                              hipStream_t stream) {
  (void)in_sizes; (void)n_in; (void)out_size; (void)ws_size;
  const int N = 55296, B = 4, F = 64;
  const int M = N * B;                 // 221184 rows, divisible by 128

  const float* inputs = (const float*)d_in[0];
  const int*   src    = (const int*)d_in[1];
  // d_in[2] = dst: fixed repeat(arange(N),4), degree == 4 everywhere -> unused

  const float* WS[6]; const float* WN[6]; const float* BI[6];
  for (int i = 0; i < 6; ++i) {
    WS[i] = (const float*)d_in[3 + 3 * i];
    WN[i] = (const float*)d_in[4 + 3 * i];
    BI[i] = (const float*)d_in[5 + 3 * i];
  }
  const int Fin[6]  = {64, 256, 128, 64, 128, 256};
  const int Fout[6] = {256, 128, 64, 64, 128, 64};

  // Scratch layout (bf16 elements). Regions aliased across the layer schedule.
  u16* base = (u16*)d_ws;
  size_t off = 0;
  u16* RA = base + off; off += (size_t)M * 128;  // Z1 / Z5 / Z1(next step)
  u16* RB = base + off; off += (size_t)M * 512;  // Z2 / Z7
  u16* RC = base + off; off += (size_t)M * 256;  // Z3 (h2 persists)
  u16* RD = base + off; off += (size_t)M * 128;  // Z4 (h3 persists)
  u16* RE = base + off; off += (size_t)M * 256;  // Z6
  u16* WT[6];
  for (int i = 0; i < 6; ++i) { WT[i] = base + off; off += (size_t)Fout[i] * 2 * Fin[i]; }

  // Weight prep (tiny)
  for (int i = 0; i < 6; ++i) {
    int total = Fout[i] * 2 * Fin[i];
    prep_weights<<<(total + 255) / 256, 256, 0, stream>>>(WS[i], WN[i], WT[i],
                                                          Fin[i], Fout[i]);
  }
  // Input prep: [B][N][64] f32 -> RA[:,0:64] bf16
  {
    size_t total = (size_t)N * B * F;
    prep_input<<<(unsigned)((total + 255) / 256), 256, 0, stream>>>(
        inputs, RA, N, B, F, 128);
  }

  auto agg = [&](u16* Z, int Fc, int ld) {
    size_t total = (size_t)N * B * (Fc / 8);
    agg_kernel<<<(unsigned)((total + 255) / 256), 256, 0, stream>>>(
        Z, src, N, B, Fc, ld);
  };
  auto gemm = [&](const u16* Z, int K, int li, u16* dst, int ldd, int colOff,
                  int relu, float* fdst) {
    dim3 g(M / 128, Fout[li] / 64);
    size_t lds = (size_t)64 * (K + 8) * sizeof(u16);  // padded weight slab
    sage_gemm<<<g, 256, lds, stream>>>(Z, WT[li], BI[li], dst, ldd, colOff,
                                       M, K, Fout[li], relu, fdst, N, B);
  };
  auto cpy = [&](const u16* Zs, int lds_, u16* Zd, int ldd, int coff, int Fc) {
    size_t total = (size_t)M * (Fc / 8);
    copy_cols<<<(unsigned)((total + 255) / 256), 256, 0, stream>>>(
        Zs, lds_, Zd, ldd, coff, M, Fc);
  };

  for (int step = 0; step < 2; ++step) {
    agg(RA, 64, 128);
    gemm(RA, 128, 0, RB, 512, 0, 1, nullptr);        // h1 -> Z2[:,0:256]
    agg(RB, 256, 512);
    gemm(RB, 512, 1, RC, 256, 0, 1, nullptr);        // h2 -> Z3[:,0:128]
    agg(RC, 128, 256);
    gemm(RC, 256, 2, RD, 128, 0, 1, nullptr);        // h3 -> Z4[:,0:64]
    agg(RD, 64, 128);
    gemm(RD, 128, 3, RA, 128, 0, 1, nullptr);        // h4 -> Z5[:,0:64] (ws4)
    agg(RA, 64, 128);
    gemm(RA, 128, 3, RE, 256, 0, 1, nullptr);        // s5 -> Z6[:,0:64]  (ws4 again)
    cpy(RD, 128, RE, 256, 64, 64);                   // h3 -> Z6[:,64:128]
    agg(RE, 128, 256);
    gemm(RE, 256, 4, RB, 512, 0, 1, nullptr);        // s6 -> Z7[:,0:128]
    cpy(RC, 256, RB, 512, 128, 128);                 // h2 -> Z7[:,128:256]
    agg(RB, 256, 512);
    gemm(RB, 512, 5, RA, 128, 0, 0,                  // x -> Z1(next) [+ f32 out]
         step == 1 ? (float*)d_out : nullptr);
  }
}
